// TransformerBlock_34385508172425
// MI455X (gfx1250) — matrix operand order repaired and checked by reading, not compiled, not run on hardware
//
#include <hip/hip_runtime.h>
#include <hip/hip_bf16.h>

// ---------------------------------------------------------------------------
// CDNA5 (gfx1250) transformer block. All matmuls via v_wmma_f32_16x16x32_f16
// (wave32, 16x16 tiles, K=32). fp32 accumulation; LN/softmax/residual in fp32.
// Global->LDS staging uses GLOBAL_LOAD_ASYNC_TO_LDS_B128 (ASYNCcnt) when the
// toolchain exposes the builtin; DPP16 butterflies for softmax reductions.
// ---------------------------------------------------------------------------

typedef __attribute__((ext_vector_type(16))) _Float16 v16h;
typedef __attribute__((ext_vector_type(8)))  float    v8f;
typedef int v4i __attribute__((vector_size(16)));   // matches async-LDS builtin

union F16Frag {            // one WMMA 16-half operand = 32 bytes
    v16h v;
    uint4 q[2];
};
union H8  { uint4 q;    _Float16 h[8];  };
union H32 { uint4 q[4]; _Float16 h[32]; };

__device__ __forceinline__ v8f wmma16(v16h a, v16h b, v8f c) {
    return __builtin_amdgcn_wmma_f32_16x16x32_f16(
        /*neg_a=*/false, a, /*neg_b=*/false, b,
        /*c_mod=*/(short)0, c, /*reuse_a=*/false, /*reuse_b=*/false);
}

__device__ __forceinline__ v8f vzero8() { v8f z = {0.f,0.f,0.f,0.f,0.f,0.f,0.f,0.f}; return z; }

// ---------------------------------------------------------------------------
// DPP16 butterfly reductions (within each 16-lane half-group, wave32).
// xor1 = quad_perm[1,0,3,2]=0xB1, xor2 = quad_perm[2,3,0,1]=0x4E,
// then row_half_mirror (0x141) and row_mirror (0x140) finish 8/16 once the
// smaller groups are uniform. No LDS traffic (vs ds_bpermute from __shfl_xor).
// ---------------------------------------------------------------------------
template <int CTRL>
__device__ __forceinline__ float dppf(float x) {
    return __int_as_float(
        __builtin_amdgcn_update_dpp(0, __float_as_int(x), CTRL, 0xf, 0xf, true));
}
__device__ __forceinline__ float rowmax16(float v) {
    v = fmaxf(v, dppf<0xB1>(v));
    v = fmaxf(v, dppf<0x4E>(v));
    v = fmaxf(v, dppf<0x141>(v));
    v = fmaxf(v, dppf<0x140>(v));
    return v;
}
__device__ __forceinline__ float rowsum16(float v) {
    v += dppf<0xB1>(v);
    v += dppf<0x4E>(v);
    v += dppf<0x141>(v);
    v += dppf<0x140>(v);
    return v;
}

// ---------------------------------------------------------------------------
// Async global->LDS copy (GLOBAL_LOAD_ASYNC_TO_LDS_B128, ASYNCcnt-tracked).
// Builtin signature (per hipcc diagnostic): (v4i AS1*, v4i AS3*, Ii, Ii).
// Generic->AS pointer conversion via integer cast: LDS generic address low
// 32 bits are the LDS offset (aperture mapping), global generic == AS1.
// ---------------------------------------------------------------------------
#if defined(__gfx1250__) && __has_builtin(__builtin_amdgcn_global_load_async_to_lds_b128)
#define HAVE_ASYNC_LDS 1
#else
#define HAVE_ASYNC_LDS 0
#endif

__device__ __forceinline__ void async_b128(const void* g, void* l) {
#if HAVE_ASYNC_LDS
    __builtin_amdgcn_global_load_async_to_lds_b128(
        (__attribute__((address_space(1))) v4i*)(uintptr_t)g,
        (__attribute__((address_space(3))) v4i*)(uint32_t)(uintptr_t)l,
        0, 0);
#else
    *(uint4*)l = *(const uint4*)g;
#endif
}

__device__ __forceinline__ void wait_async0() {
#if HAVE_ASYNC_LDS
#if __has_builtin(__builtin_amdgcn_s_wait_asynccnt)
    __builtin_amdgcn_s_wait_asynccnt(0);
#else
    asm volatile("s_wait_asynccnt 0x0" ::: "memory");
#endif
#endif
}

// ---------------------------------------------------------------------------
// Weight repack kernels (fp32 -> f16, plus head-blocked QKV concat)
// ---------------------------------------------------------------------------
__global__ void pack_qkv_w_kernel(const float* __restrict__ Wq,
                                  const float* __restrict__ Wk,
                                  const float* __restrict__ Wv,
                                  _Float16* __restrict__ W) {
    // W[d*2304 + which*768 + h*64 + e] = Wsrc[h*768*64 + d*64 + e]
    const int total = 768 * 2304;
    for (int idx = blockIdx.x * blockDim.x + threadIdx.x; idx < total;
         idx += gridDim.x * blockDim.x) {
        int d = idx / 2304, c = idx % 2304;
        int which = c / 768, cc = c % 768;
        int h = cc / 64, e = cc % 64;
        const float* src = (which == 0) ? Wq : (which == 1) ? Wk : Wv;
        W[idx] = (_Float16)src[(size_t)h * 768 * 64 + (size_t)d * 64 + e];
    }
}

__global__ void pack_bias_qkv_kernel(const float* __restrict__ bq,
                                     const float* __restrict__ bk,
                                     const float* __restrict__ bv,
                                     float* __restrict__ o) {
    int i = blockIdx.x * blockDim.x + threadIdx.x;
    if (i < 2304) o[i] = (i < 768) ? bq[i] : (i < 1536) ? bk[i - 768] : bv[i - 1536];
}

__global__ void f32_to_f16_kernel(const float* __restrict__ s,
                                  _Float16* __restrict__ d, int n) {
    for (int i = blockIdx.x * blockDim.x + threadIdx.x; i < n;
         i += gridDim.x * blockDim.x) d[i] = (_Float16)s[i];
}

// ---------------------------------------------------------------------------
// Fused LayerNorm -> f16 activation. One 256-thread block per row, D=768.
// ---------------------------------------------------------------------------
__global__ __launch_bounds__(256)
void ln_f16_kernel(const float* __restrict__ x, const float* __restrict__ g,
                   const float* __restrict__ b, _Float16* __restrict__ out, int D) {
    const int row = blockIdx.x;
    const float* xr = x + (size_t)row * D;
    __shared__ float red[8];
    const int tid = threadIdx.x;

    float s = 0.f;
    for (int c = tid; c < D; c += 256) s += xr[c];
    s = rowsum16(s);
    s += __shfl_xor(s, 16, 32);
    if ((tid & 31) == 0) red[tid >> 5] = s;
    __syncthreads();
    float mu = 0.f;
#pragma unroll
    for (int i = 0; i < 8; ++i) mu += red[i];
    mu /= (float)D;
    __syncthreads();

    float v = 0.f;
    for (int c = tid; c < D; c += 256) { float d = xr[c] - mu; v += d * d; }
    v = rowsum16(v);
    v += __shfl_xor(v, 16, 32);
    if ((tid & 31) == 0) red[tid >> 5] = v;
    __syncthreads();
    float var = 0.f;
#pragma unroll
    for (int i = 0; i < 8; ++i) var += red[i];
    var /= (float)D;
    const float rs = rsqrtf(var + 1e-5f);

    for (int c = tid; c < D; c += 256)
        out[(size_t)row * D + c] = (_Float16)((xr[c] - mu) * rs * g[c] + b[c]);
}

// ---------------------------------------------------------------------------
// Tiled WMMA GEMM: C[M,N] = A[M,K](f16) * B[K,N](f16) + bias, opt ReLU,
// opt residual add (fp32), outputs fp32 and/or f16.
// BM=256, BN=64, BK=32; 256 threads = 8 waves; each wave -> 32x64 (8 WMMA/iter)
// A tile staged via async global->LDS; B tile transposed via VGPR scatter.
// ---------------------------------------------------------------------------
template <bool RELU, bool RESID, bool OUTF, bool OUTH>
__global__ __launch_bounds__(256)
void gemm_f16_kernel(const _Float16* __restrict__ A, const _Float16* __restrict__ Bm,
                     const float* __restrict__ bias, const float* __restrict__ resid,
                     float* __restrict__ Cf, _Float16* __restrict__ Ch,
                     int M, int N, int K) {
    const int BM = 256, BN = 64, BK = 32;
    __shared__ __align__(16) _Float16 As[BM * BK];   // 16 KB, row-major [m][k]
    __shared__ __align__(16) _Float16 Bt[BN * BK];   // 4 KB, transposed [n][k]

    const int tid  = threadIdx.x;
    const int lane = tid & 31;
    const int w    = tid >> 5;
    const int lh   = lane & 15;
    const int kbA  = (lane >> 4) * 8;    // A-frag K sub-offset (interleaved)
    const int kbB  = (lane >> 4) * 8;    // B-frag K sub-offset (interleaved)
    const int blockM = blockIdx.y * BM;
    const int blockN = blockIdx.x * BN;

    v8f acc[2][4];
#pragma unroll
    for (int i = 0; i < 2; ++i)
#pragma unroll
        for (int j = 0; j < 4; ++j) acc[i][j] = vzero8();

    for (int kk = 0; kk < K; kk += BK) {
        __syncthreads();
        {   // stage A: one 32-half row per thread, 4 async b128 (512 B/wave/op)
            const _Float16* gp = A + (size_t)(blockM + tid) * K + kk;
            _Float16* sp = As + tid * BK;
#pragma unroll
            for (int i = 0; i < 4; ++i) async_b128(gp + 8 * i, sp + 8 * i);
        }
        {   // stage B transposed: Bt[n][k] (overlaps with async A)
            int kr = tid >> 3;
            int n0 = (tid & 7) * 8;
            H8 bvv;
            bvv.q = *(const uint4*)(Bm + (size_t)(kk + kr) * N + blockN + n0);
#pragma unroll
            for (int i = 0; i < 8; ++i) Bt[(n0 + i) * BK + kr] = bvv.h[i];
        }
        wait_async0();
        __syncthreads();

        F16Frag af[2], bf[4];
#pragma unroll
        for (int mi = 0; mi < 2; ++mi) {
            const _Float16* p = As + (32 * w + 16 * mi + lh) * BK + kbA;
            af[mi].q[0] = *(const uint4*)p;
            af[mi].q[1] = *(const uint4*)(p + 16);
        }
#pragma unroll
        for (int nt = 0; nt < 4; ++nt) {
            const _Float16* p = Bt + (16 * nt + lh) * BK + kbB;
            bf[nt].q[0] = *(const uint4*)p;
            bf[nt].q[1] = *(const uint4*)(p + 16);
        }
#pragma unroll
        for (int mi = 0; mi < 2; ++mi)
#pragma unroll
            for (int nt = 0; nt < 4; ++nt)
                acc[mi][nt] = wmma16(af[mi].v, bf[nt].v, acc[mi][nt]);
    }

    // epilogue: C layout => lane holds col lh, rows r + 8*(lane/16)
#pragma unroll
    for (int mi = 0; mi < 2; ++mi)
#pragma unroll
        for (int nt = 0; nt < 4; ++nt)
#pragma unroll
            for (int r = 0; r < 8; ++r) {
                int row = blockM + 32 * w + 16 * mi + r + 8 * (lane >> 4);
                int col = blockN + 16 * nt + lh;
                float v = acc[mi][nt][r] + bias[col];
                if (RELU)  v = v > 0.f ? v : 0.f;
                if (RESID) v += resid[(size_t)row * N + col];
                if (OUTF)  Cf[(size_t)row * N + col] = v;
                if (OUTH)  Ch[(size_t)row * N + col] = (_Float16)v;
            }
}

// ---------------------------------------------------------------------------
// Flash-style attention. Block = 256 thr (8 waves), handles 128 query rows of
// one (b,h); K/V tiles staged once per 128 rows. Online softmax in fp32 with
// DPP16 row reductions; scores & PV via WMMA (16 per key-tile per wave).
// qkv: f16 [4096 x 2304] (Q|K|V head-blocked). attn out: f16.
// ---------------------------------------------------------------------------
__global__ __launch_bounds__(256)
void attn_kernel(const _Float16* __restrict__ qkv, _Float16* __restrict__ attn) {
    const int S = 2048, NC = 2304, H = 12;
    const float scale = 0.125f;           // 1/sqrt(64)

    const int bh = blockIdx.y;
    const int b = bh / H, h = bh % H;
    const int q0 = blockIdx.x * 128;
    const int tid = threadIdx.x;
    const int lane = tid & 31;
    const int w = tid >> 5;
    const int lh = lane & 15;
    const int kbA = (lane >> 4) * 8;
    const int kbB = (lane >> 4) * 8;
    const size_t rowbase = (size_t)b * S;

    __shared__ __align__(16) _Float16 Ks[64 * 64];       // K tile [key][feat]  8 KB
    __shared__ __align__(16) _Float16 Vt[64 * 64];       // V tile [feat][key]  8 KB
    __shared__ __align__(16) _Float16 Ps[8 * 16 * 64];   // per-wave P tiles   16 KB

    // load Q fragments for this wave's 16 rows (A layout, two K=32 chunks)
    F16Frag qf[2];
    {
        const _Float16* qp = qkv + (rowbase + q0 + 16 * w + lh) * NC + h * 64;
#pragma unroll
        for (int c = 0; c < 2; ++c) {
            qf[c].q[0] = *(const uint4*)(qp + 32 * c + kbA);
            qf[c].q[1] = *(const uint4*)(qp + 32 * c + 16 + kbA);
        }
    }

    v8f oacc[4];
#pragma unroll
    for (int t = 0; t < 4; ++t) oacc[t] = vzero8();
    float mrow[8], lrow[8];
#pragma unroll
    for (int r = 0; r < 8; ++r) { mrow[r] = -3.0e38f; lrow[r] = 0.f; }

    for (int j = 0; j < S; j += 64) {
        __syncthreads();
        {   // stage K (async, row-major) and V (transposed via VGPR scatter)
            int kl = tid >> 2;            // 0..63
            int e0 = (tid & 3) * 16;      // 0,16,32,48
            const _Float16* kg = qkv + (rowbase + j + kl) * NC + 768 + h * 64 + e0;
            _Float16* ks = Ks + kl * 64 + e0;
            async_b128(kg, ks);
            async_b128(kg + 8, ks + 8);
            H8 v0, v1;
            const _Float16* vg = qkv + (rowbase + j + kl) * NC + 1536 + h * 64 + e0;
            v0.q = *(const uint4*)vg;
            v1.q = *(const uint4*)(vg + 8);
#pragma unroll
            for (int i = 0; i < 8; ++i) {
                Vt[(e0 + i) * 64 + kl]     = v0.h[i];
                Vt[(e0 + 8 + i) * 64 + kl] = v1.h[i];
            }
        }
        wait_async0();
        __syncthreads();

        // scores: S = Q @ K^T  -> 4 col-tiles x 2 K-chunks
        v8f sacc[4];
#pragma unroll
        for (int t = 0; t < 4; ++t) sacc[t] = vzero8();
#pragma unroll
        for (int nt = 0; nt < 4; ++nt)
#pragma unroll
            for (int c = 0; c < 2; ++c) {
                F16Frag kf;
                const _Float16* p = Ks + (16 * nt + lh) * 64 + 32 * c + kbB;
                kf.q[0] = *(const uint4*)p;
                kf.q[1] = *(const uint4*)(p + 16);
                sacc[nt] = wmma16(qf[c].v, kf.v, sacc[nt]);
            }

        // online softmax: per-row max/sum via DPP16 butterflies (no LDS)
        float mnew[8];
#pragma unroll
        for (int r = 0; r < 8; ++r) {
            float v = -3.0e38f;
#pragma unroll
            for (int t = 0; t < 4; ++t) {
                float sv = sacc[t][r] * scale;
                sacc[t][r] = sv;
                v = fmaxf(v, sv);
            }
            v = rowmax16(v);
            mnew[r] = fmaxf(mrow[r], v);
        }
#pragma unroll
        for (int r = 0; r < 8; ++r) {
            float cf = __expf(mrow[r] - mnew[r]);
            lrow[r] *= cf;
#pragma unroll
            for (int t = 0; t < 4; ++t) oacc[t][r] *= cf;
            mrow[r] = mnew[r];
            float rs = 0.f;
#pragma unroll
            for (int t = 0; t < 4; ++t) {
                float p = __expf(sacc[t][r] - mnew[r]);
                sacc[t][r] = p;
                rs += p;
            }
            rs = rowsum16(rs);
            lrow[r] += rs;
        }

        // restage P (C layout -> LDS row-major) for use as WMMA A operand
        _Float16* pw = Ps + w * 1024;
#pragma unroll
        for (int t = 0; t < 4; ++t)
#pragma unroll
            for (int r = 0; r < 8; ++r)
                pw[(r + 8 * (lane >> 4)) * 64 + 16 * t + lh] = (_Float16)sacc[t][r];
        __syncthreads();

        // O += P @ V
#pragma unroll
        for (int c = 0; c < 2; ++c) {
            F16Frag pf;
            const _Float16* pp = pw + lh * 64 + 32 * c + kbA;
            pf.q[0] = *(const uint4*)pp;
            pf.q[1] = *(const uint4*)(pp + 16);
#pragma unroll
            for (int nt = 0; nt < 4; ++nt) {
                F16Frag vf;
                const _Float16* vp = Vt + (16 * nt + lh) * 64 + 32 * c + kbB;
                vf.q[0] = *(const uint4*)vp;
                vf.q[1] = *(const uint4*)(vp + 16);
                oacc[nt] = wmma16(pf.v, vf.v, oacc[nt]);
            }
        }
    }

    // normalize and write concat-head output (f16, feeds Wo GEMM)
#pragma unroll
    for (int t = 0; t < 4; ++t)
#pragma unroll
        for (int r = 0; r < 8; ++r) {
            size_t row = rowbase + q0 + 16 * w + r + 8 * (lane >> 4);
            attn[row * 768 + h * 64 + 16 * t + lh] = (_Float16)(oacc[t][r] / lrow[r]);
        }
}

// ---------------------------------------------------------------------------
// Launch: LN1 -> QKV GEMM -> attention -> Wo GEMM(+resid) -> LN2 ->
//         FF1 GEMM(ReLU) -> FF2 GEMM(+resid -> d_out)
// ---------------------------------------------------------------------------
extern "C" void kernel_launch(void* const* d_in, const int* in_sizes, int n_in,
                              void* d_out, int out_size, void* d_ws, size_t ws_size,
                              hipStream_t stream) {
    (void)in_sizes; (void)n_in; (void)out_size; (void)ws_size;
    const int B = 2, S = 2048, D = 768, HD4 = 3072, NC = 2304;
    const int M = B * S;  // 4096 token rows

    const float* x   = (const float*)d_in[0];
    const float* Wq  = (const float*)d_in[1];
    const float* bq  = (const float*)d_in[2];
    const float* Wk  = (const float*)d_in[3];
    const float* bk  = (const float*)d_in[4];
    const float* Wv  = (const float*)d_in[5];
    const float* bv  = (const float*)d_in[6];
    const float* Wo  = (const float*)d_in[7];
    const float* bo  = (const float*)d_in[8];
    const float* g1  = (const float*)d_in[9];
    const float* b1  = (const float*)d_in[10];
    const float* g2  = (const float*)d_in[11];
    const float* b2  = (const float*)d_in[12];
    const float* W1  = (const float*)d_in[13];
    const float* bf1 = (const float*)d_in[14];
    const float* W2  = (const float*)d_in[15];
    const float* bf2 = (const float*)d_in[16];
    float* out = (float*)d_out;

    // workspace carve-out (~90 MB), 256-byte aligned slices
    char* wsb = (char*)d_ws;
    size_t off = 0;
    auto alloc = [&](size_t bytes) -> char* {
        char* p = wsb + off;
        off = (off + bytes + 255) & ~(size_t)255;
        return p;
    };
    _Float16* xn    = (_Float16*)alloc((size_t)M * D * 2);
    _Float16* xn2   = (_Float16*)alloc((size_t)M * D * 2);
    _Float16* qkv   = (_Float16*)alloc((size_t)M * NC * 2);
    _Float16* attnb = (_Float16*)alloc((size_t)M * D * 2);
    _Float16* hbuf  = (_Float16*)alloc((size_t)M * HD4 * 2);
    float*    x1    = (float*)   alloc((size_t)M * D * 4);
    _Float16* Wqkvh = (_Float16*)alloc((size_t)D * NC * 2);
    _Float16* Woh   = (_Float16*)alloc((size_t)D * D * 2);
    _Float16* W1h   = (_Float16*)alloc((size_t)D * HD4 * 2);
    _Float16* W2h   = (_Float16*)alloc((size_t)HD4 * D * 2);
    float*    bqkv  = (float*)   alloc((size_t)NC * 4);

    // --- weight repack (f32 -> f16) ---
    pack_qkv_w_kernel<<<dim3((D * NC + 255) / 256), 256, 0, stream>>>(Wq, Wk, Wv, Wqkvh);
    pack_bias_qkv_kernel<<<dim3((NC + 255) / 256), 256, 0, stream>>>(bq, bk, bv, bqkv);
    f32_to_f16_kernel<<<dim3(2048), 256, 0, stream>>>(Wo, Woh, D * D);
    f32_to_f16_kernel<<<dim3(4096), 256, 0, stream>>>(W1, W1h, D * HD4);
    f32_to_f16_kernel<<<dim3(4096), 256, 0, stream>>>(W2, W2h, HD4 * D);

    // --- LN1 ---
    ln_f16_kernel<<<dim3(M), 256, 0, stream>>>(x, g1, b1, xn, D);

    // --- QKV projection: [4096 x 768] @ [768 x 2304] -> f16 qkv ---
    gemm_f16_kernel<false, false, false, true>
        <<<dim3(NC / 64, M / 256), 256, 0, stream>>>(
            xn, Wqkvh, bqkv, nullptr, nullptr, qkv, M, NC, D);

    // --- attention: 128 query rows per block, one (b,h) per blockIdx.y ---
    attn_kernel<<<dim3(S / 128, B * 12), 256, 0, stream>>>(qkv, attnb);

    // --- output projection + residual: x1 = x + attn @ Wo + bo ---
    gemm_f16_kernel<false, true, true, false>
        <<<dim3(D / 64, M / 256), 256, 0, stream>>>(
            attnb, Woh, bo, x, x1, nullptr, M, D, D);

    // --- LN2 ---
    ln_f16_kernel<<<dim3(M), 256, 0, stream>>>(x1, g2, b2, xn2, D);

    // --- FF1: relu(xn2 @ W1 + bf1) -> f16 h ---
    gemm_f16_kernel<true, false, false, true>
        <<<dim3(HD4 / 64, M / 256), 256, 0, stream>>>(
            xn2, W1h, bf1, nullptr, nullptr, hbuf, M, HD4, D);

    // --- FF2: out = x1 + h @ W2 + bf2 ---
    gemm_f16_kernel<false, true, true, false>
        <<<dim3(D / 64, M / 256), 256, 0, stream>>>(
            hbuf, W2h, bf2, x1, out, nullptr, M, D, HD4);
}
